// RecursiveNN_94489281191
// MI455X (gfx1250) — matrix operand order repaired and checked by reading, not compiled, not run on hardware
//
#include <hip/hip_runtime.h>
#include <hip/hip_bf16.h>
#include <stdint.h>

// ---------------------------------------------------------------------------
// RecursiveNN on MI455X (gfx1250):
//   leaves  : relu(emb[words])           -> bf16 states
//   compose : 9 levels of GEMM+ReLU      -> v_wmma_f32_16x16x32_bf16,
//             tiles staged with GLOBAL_LOAD_ASYNC_TO_LDS_B128 (ASYNCcnt),
//             double-buffered LDS, one barrier per K-step.
//   project : states @ P^T + pb          -> fp32 out
// Precision: bf16 inputs / fp32 accumulation (CDNA5 WMMA sweet spot);
// bf16 intermediates halve HBM traffic (~0.5 GB total @ 23.3 TB/s).
// ---------------------------------------------------------------------------

typedef __attribute__((ext_vector_type(16))) __bf16 v16bf;
typedef __attribute__((ext_vector_type(8)))  float  v8f;

#define EDIM   256
#define LDIM   512
#define BDIM   256
#define NODES  1023          // 2*L - 1
#define CDIM   5

#define TM 128               // block tile M (8 waves x 16)
#define TN 64                // block tile N (4 wmma strips)
#define TK 32                // wmma K per stage
#define LDA_STRIDE 40        // ushorts per LDS row (32 + 8 pad), 16B-aligned rows
#define LDW_STRIDE 40

__device__ __forceinline__ unsigned short f2bf(float x) {
    unsigned int u = __float_as_uint(x);
    u += 0x7FFFu + ((u >> 16) & 1u);         // round-to-nearest-even
    return (unsigned short)(u >> 16);
}
__device__ __forceinline__ float bf2f(unsigned int h) {
    return __uint_as_float(h << 16);
}

// Async global->LDS b128 (gfx1250). vdst = LDS byte offset, vaddr = 64-bit VA.
// Low 32 bits of a generic pointer to __shared__ are the LDS byte offset
// (flat->LDS mapping truncates to addr[31:0] per CDNA5 ISA 10.2).
__device__ __forceinline__ void async_b128(unsigned lds_off,
                                           unsigned long long gaddr) {
    asm volatile("global_load_async_to_lds_b128 %0, %1, off"
                 :: "v"(lds_off), "v"(gaddr) : "memory");
}
__device__ __forceinline__ void wait_async0() {
#if defined(__has_builtin) && __has_builtin(__builtin_amdgcn_s_wait_asynccnt)
    __builtin_amdgcn_s_wait_asynccnt(0);
#else
    asm volatile("s_wait_asynccnt 0" ::: "memory");
#endif
}

// ---- W fp32 -> bf16 (layout [e][k] is already WMMA-B friendly) -------------
__global__ void cvt_w_kernel(const float* __restrict__ Wf,
                             unsigned short* __restrict__ Wb, int n) {
    int i = blockIdx.x * blockDim.x + threadIdx.x;
    if (i < n) Wb[i] = f2bf(Wf[i]);
}

// ---- leaves: relu(emb[word]) -> states[b][l][*] ---------------------------
__global__ __launch_bounds__(EDIM) void leaf_kernel(
        const int* __restrict__ words, const float* __restrict__ emb,
        unsigned short* __restrict__ states) {
    const int row = blockIdx.x;              // b*512 + l
    const int b = row >> 9, l = row & 511;
    const int word = words[row];
    float v = emb[(size_t)word * EDIM + threadIdx.x];
    v = v > 0.f ? v : 0.f;
    states[((size_t)(b * NODES + l)) * EDIM + threadIdx.x] = f2bf(v);
}

// ---- compose level: H_out = relu(A(Mx512) x W^T + b), A = concat'd pairs --
__global__ __launch_bounds__(256) void compose_kernel(
        unsigned short* __restrict__ states,
        const unsigned short* __restrict__ Wb,
        const float* __restrict__ bias,
        int in_off, int out_off, int ppb_log2) {
    __shared__ __align__(16) unsigned short lA[2][TM * LDA_STRIDE]; // 2x10240 B
    __shared__ __align__(16) unsigned short lW[2][TN * LDW_STRIDE]; // 2x 5120 B

    const int tid  = threadIdx.x;
    const int lane = tid & 31;
    const int wave = tid >> 5;
    const int half = lane >> 4;      // 0: lanes 0-15, 1: lanes 16-31
    const int l16  = lane & 15;
    const int mBlock = blockIdx.x;
    const int nbase  = blockIdx.y * TN;
    const int ppbm1  = (1 << ppb_log2) - 1;

    // ---- per-thread staging roles: 2 A rows + 1 W row, one b128 each ------
    const int seg  = tid & 3;                 // 16B segment within a 64B row
    const int rowT = tid >> 2;                // 0..63
    int  rA[2] = { rowT, rowT + 64 };
    unsigned long long gA[2];
#pragma unroll
    for (int rep = 0; rep < 2; ++rep) {
        const int gr = mBlock * TM + rA[rep];
        const int bb = gr >> ppb_log2;
        const int ii = gr & ppbm1;
        gA[rep] = (unsigned long long)(uintptr_t)
            (states + ((size_t)(bb * NODES + in_off + 2 * ii)) * EDIM + seg * 8);
    }
    unsigned long long gW = (unsigned long long)(uintptr_t)
        (Wb + ((size_t)(nbase + rowT)) * (2 * EDIM) + seg * 8);

    unsigned ldsA[2][2], ldsW[2];
#pragma unroll
    for (int c = 0; c < 2; ++c) {
#pragma unroll
        for (int rep = 0; rep < 2; ++rep)
            ldsA[c][rep] = (unsigned)(uintptr_t)
                &lA[c][rA[rep] * LDA_STRIDE + seg * 8];
        ldsW[c] = (unsigned)(uintptr_t)&lW[c][rowT * LDW_STRIDE + seg * 8];
    }

    v8f acc[4] = {};
    const int am = wave * 16 + l16;

    // Prologue: stage tile 0 asynchronously.
    async_b128(ldsA[0][0], gA[0]);
    async_b128(ldsA[0][1], gA[1]);
    async_b128(ldsW[0],    gW);

    for (int kt = 0; kt < 512 / TK; ++kt) {
        const int cur = kt & 1;
        wait_async0();          // this wave's tile data is in LDS
        __syncthreads();        // everyone's tile data is in LDS

        if (kt < 512 / TK - 1) {
            // Issue next tile into the other buffer; overlaps with WMMA below.
            gA[0] += TK * 2; gA[1] += TK * 2; gW += TK * 2;   // +64 bytes
            const int nxt = cur ^ 1;
            async_b128(ldsA[nxt][0], gA[0]);
            async_b128(ldsA[nxt][1], gA[1]);
            async_b128(ldsW[nxt],    gW);
        }

        // A fragment (ISA 16-bit A 16x32 layout): lanes 0-15 take K 0-7,16-23;
        // lanes 16-31 take K 8-15,24-31. Each half = two 16B LDS reads.
        union { v16bf v; uint4 q[2]; } afrag;
        const unsigned short* la = lA[cur];
        afrag.q[0] = *(const uint4*)&la[am * LDA_STRIDE + half * 8];
        afrag.q[1] = *(const uint4*)&la[am * LDA_STRIDE + half * 8 + 16];

        // 4 B fragments (N strips), one A fragment reused -> 4 WMMA.
        const unsigned short* lw = lW[cur];
#pragma unroll
        for (int s = 0; s < 4; ++s) {
            union { v16bf v; uint4 q[2]; } bfrag;
            const int be = s * 16 + l16;    // B row = output column e
            bfrag.q[0] = *(const uint4*)&lw[be * LDW_STRIDE + half * 8];
            bfrag.q[1] = *(const uint4*)&lw[be * LDW_STRIDE + half * 8 + 16];
            acc[s] = __builtin_amdgcn_wmma_f32_16x16x32_bf16(
                false, afrag.v, false, bfrag.v, (short)0, acc[s], false, false);
        }
    }

    // Epilogue: bias + ReLU + bf16 store.
    // D layout: VGPR r, lane l -> M = r + 8*(l>=16), N = l%16.
    const int grow_base = mBlock * TM + wave * 16;
#pragma unroll
    for (int s = 0; s < 4; ++s) {
        const int e  = nbase + s * 16 + l16;
        const float bv = bias[e];
#pragma unroll
        for (int r = 0; r < 8; ++r) {
            const int grow = grow_base + r + 8 * half;
            const int bb = grow >> ppb_log2;
            const int ii = grow & ppbm1;
            float v = acc[s][r] + bv;
            v = v > 0.f ? v : 0.f;
            states[((size_t)(bb * NODES + out_off + ii)) * EDIM + e] = f2bf(v);
        }
    }
}

// ---- projection: one wave per node row, wave32 shuffle reduction ----------
__global__ __launch_bounds__(256) void proj_kernel(
        const unsigned short* __restrict__ states,
        const float* __restrict__ P, const float* __restrict__ pb,
        float* __restrict__ out) {
    const int lane = threadIdx.x & 31;
    const int wave = threadIdx.x >> 5;
    const long row = (long)blockIdx.x * 8 + wave;     // b*1023 + node

    union { uint4 q; unsigned int u[4]; } d;
    d.q = *(const uint4*)&states[(size_t)row * EDIM + lane * 8];
    float x[8];
#pragma unroll
    for (int k = 0; k < 4; ++k) {
        x[2 * k]     = bf2f(d.u[k] & 0xFFFFu);
        x[2 * k + 1] = bf2f(d.u[k] >> 16);
    }

    float accv[CDIM];
#pragma unroll
    for (int c = 0; c < CDIM; ++c) {
        float a = 0.f;
#pragma unroll
        for (int j = 0; j < 8; ++j)
            a += x[j] * P[c * EDIM + lane * 8 + j];
#pragma unroll
        for (int off = 16; off > 0; off >>= 1)
            a += __shfl_xor(a, off, 32);
        accv[c] = a;
    }
    if (lane == 0) {
#pragma unroll
        for (int c = 0; c < CDIM; ++c)
            out[row * CDIM + c] = accv[c] + pb[c];
    }
}

// ---------------------------------------------------------------------------
extern "C" void kernel_launch(void* const* d_in, const int* in_sizes, int n_in,
                              void* d_out, int out_size, void* d_ws, size_t ws_size,
                              hipStream_t stream) {
    const int*   words = (const int*)d_in[0];
    const float* emb   = (const float*)d_in[1];
    const float* Wf    = (const float*)d_in[2];
    const float* bias  = (const float*)d_in[3];
    const float* P     = (const float*)d_in[4];
    const float* pb    = (const float*)d_in[5];
    float*       out   = (float*)d_out;

    // Workspace: bf16 states (B,1023,E) ~134 MB, then bf16 W (256 KB).
    unsigned short* states = (unsigned short*)d_ws;
    const size_t statesElems = (size_t)BDIM * NODES * EDIM;
    unsigned short* Wb = states + statesElems;

    cvt_w_kernel<<<(EDIM * 2 * EDIM + 255) / 256, 256, 0, stream>>>(
        Wf, Wb, EDIM * 2 * EDIM);

    leaf_kernel<<<BDIM * LDIM, EDIM, 0, stream>>>(words, emb, states);

    int in_off = 0;
    for (int j = 0; j < 9; ++j) {
        const int n        = LDIM >> j;     // children at this level
        const int ppb_log2 = 8 - j;         // log2(parents per batch)
        const int out_off  = in_off + n;
        const int M        = BDIM << ppb_log2;
        dim3 grid(M / TM, EDIM / TN);       // (512>>j, 4)
        compose_kernel<<<grid, 256, 0, stream>>>(
            states, Wb, bias, in_off, out_off, ppb_log2);
        in_off = out_off;
    }

    proj_kernel<<<(BDIM * NODES) / 8, 256, 0, stream>>>(states, P, pb, out);
}
